// CategoryConstrainedGRUMN_54975581388890
// MI455X (gfx1250) — compile-verified, matrix operand order checked by reading, not run on hardware
//
#include <hip/hip_runtime.h>

typedef unsigned short ushort_t;
typedef __attribute__((ext_vector_type(16))) __bf16 v16bf;
typedef __attribute__((ext_vector_type(8)))  float  v8f;

#define Bv   4
#define Tv   256
#define Ev   256
#define Hv   256
#define Av   256
#define Vv   32000
#define CTXN 500
#define CTXP 512

// ---------------- TDM availability ----------------
#ifdef __has_builtin
#if __has_builtin(__builtin_amdgcn_tensor_load_to_lds) && __has_builtin(__builtin_amdgcn_s_wait_tensorcnt)
#define HAVE_TDM 1
#endif
#endif
#ifndef HAVE_TDM
#define HAVE_TDM 0
#endif

typedef __attribute__((ext_vector_type(4))) unsigned int tdm_u32x4;
typedef __attribute__((ext_vector_type(8))) int          tdm_i32x8;
typedef __attribute__((ext_vector_type(4))) int          tdm_i32x4;

// ---------- bf16 helpers ----------
__device__ __forceinline__ ushort_t f2bf(float f) {
  unsigned int x = __builtin_bit_cast(unsigned int, f);
  unsigned int r = x + 0x7FFFu + ((x >> 16) & 1u);   // round-to-nearest-even
  return (ushort_t)(r >> 16);
}

// ---------- WMMA fragment loads (vectorized: 2x b128 per fragment) ----------
// A: row-major [16 x lda] bf16, pre-offset to k0. Per CDNA5 ISA 7.12.2:
// lanes 0-15: rows 0-15, K {0..7, 16..23}; lanes 16-31: K {8..15, 24..31}.
__device__ __forceinline__ v16bf load_a_frag(const ushort_t* __restrict__ A, int lda, int lane) {
  const int hi = (lane >> 4) & 1;
  const int mr = lane & 15;
  union { uint4 q[2]; v16bf v; } a;
  a.q[0] = *(const uint4*)(A + (size_t)mr * lda + hi * 8);
  a.q[1] = *(const uint4*)(A + (size_t)mr * lda + 16 + hi * 8);
  return a.v;
}
// B pre-packed into fragment layout: [tile][lane][16] halves, contiguous per lane.
__device__ __forceinline__ v16bf load_b_frag(const ushort_t* __restrict__ Bfrag, int lane) {
  union { uint4 q[2]; v16bf v; } b;
  const ushort_t* bp = Bfrag + lane * 16;
  b.q[0] = *(const uint4*)bp;
  b.q[1] = *(const uint4*)(bp + 8);
  return b.v;
}

// ---------- pack f32 [sR,sC] -> bf16 WMMA-B fragments [Kp/32][Np/16][32][16] ----------
__global__ void k_pack_b(const float* __restrict__ src, ushort_t* __restrict__ dst,
                         int sR, int sC, int Kp, int Np) {
  const int ntN = Np >> 4;
  long total = (long)(Kp >> 5) * ntN * 512;
  for (long o = (long)blockIdx.x * blockDim.x + threadIdx.x; o < total;
       o += (long)gridDim.x * blockDim.x) {
    int i    = (int)(o & 15);
    int lane = (int)((o >> 4) & 31);
    long tile = o >> 9;
    int tn = (int)(tile % ntN);
    int tk = (int)(tile / ntN);
    int hi = lane >> 4;
    int n  = tn * 16 + (lane & 15);
    int k  = tk * 32 + ((i < 8) ? (hi * 8 + i) : (16 + hi * 8 + (i - 8)));
    float v = (k < sR && n < sC) ? src[(long)k * sC + n] : 0.0f;
    dst[o] = f2bf(v);
  }
}

// ---------- embedding gather + layer-0 input: li0 = relu(concat(we,ce)+tile(ce,2)) ----------
__global__ void k_prep(const int* __restrict__ ids, const int* __restrict__ cids,
                       const float* __restrict__ emb, const float* __restrict__ cemb,
                       ushort_t* __restrict__ liA) {
  long n = (long)Bv * Tv * 512;
  for (long i = (long)blockIdx.x * blockDim.x + threadIdx.x; i < n;
       i += (long)gridDim.x * blockDim.x) {
    int c = (int)(i & 511);
    long bt = i >> 9;
    int wid = ids[bt], cid = cids[bt];
    float ce = cemb[(long)cid * Ev + (c & 255)];
    float x  = (c < 256) ? emb[(long)wid * Ev + c] : cemb[(long)cid * Ev + (c - 256)];
    liA[i] = f2bf(fmaxf(x + ce, 0.0f));
  }
}

// ---------- big bf16 WMMA GEMM, 4-wide N blocking: out[M,N]=A[M,K]*B+bias ----------
__global__ void __launch_bounds__(256)
k_gemm_bf16(const ushort_t* __restrict__ A, const ushort_t* __restrict__ Bp,
            const float* __restrict__ bias, float* __restrict__ out,
            int M, int N, int K, int lda, int ldo, int biasN) {
  const int lane = threadIdx.x & 31;
  const int wave = threadIdx.x >> 5;
  const int ntN  = N >> 4;
  const int ntG  = ntN >> 2;                 // groups of 4 column tiles
  const long unit = (long)blockIdx.x * (blockDim.x >> 5) + wave;
  const int tm  = (int)(unit / ntG);
  const int tng = (int)(unit % ntG);
  if (tm >= (M >> 4)) return;
  const ushort_t* Ab = A + (size_t)tm * 16 * lda;
  const int tn0 = tng * 4;
  v8f acc0 = {}, acc1 = {}, acc2 = {}, acc3 = {};
  const int ktiles = K >> 5;
  for (int tk = 0; tk < ktiles; ++tk) {
    const ushort_t* Brow = Bp + ((size_t)tk * ntN + tn0) * 512;
    __builtin_prefetch(Brow + (size_t)ntN * 512, 0, 1);
    v16bf a = load_a_frag(Ab + tk * 32, lda, lane);
    acc0 = __builtin_amdgcn_wmma_f32_16x16x32_bf16(false, a, false, load_b_frag(Brow,            lane), (short)0, acc0, false, false);
    acc1 = __builtin_amdgcn_wmma_f32_16x16x32_bf16(false, a, false, load_b_frag(Brow + 512,      lane), (short)0, acc1, false, false);
    acc2 = __builtin_amdgcn_wmma_f32_16x16x32_bf16(false, a, false, load_b_frag(Brow + 2 * 512,  lane), (short)0, acc2, false, false);
    acc3 = __builtin_amdgcn_wmma_f32_16x16x32_bf16(false, a, false, load_b_frag(Brow + 3 * 512,  lane), (short)0, acc3, false, false);
  }
  const int hi = (lane >> 4) & 1;
  v8f* accs[4] = { &acc0, &acc1, &acc2, &acc3 };
#pragma unroll
  for (int j = 0; j < 4; ++j) {
    const int col = (tn0 + j) * 16 + (lane & 15);
    const float bval = (bias != nullptr && col < biasN) ? bias[col] : 0.0f;
#pragma unroll
    for (int r = 0; r < 8; ++r) {
      int m = r + hi * 8;
      out[(size_t)(tm * 16 + m) * ldo + col] = (*accs[j])[r] + bval;
    }
  }
}

// ---------- in-scan wave-cooperative GEMM (A in LDS, 16 padded rows, 4 valid) ----------
__device__ __forceinline__ void scan_gemm(const ushort_t* __restrict__ A, int lda,
                                          const ushort_t* __restrict__ Bp,
                                          int N, int K,
                                          const float* __restrict__ bias, int biasN,
                                          float* __restrict__ out, int rowStride,
                                          int tid) {
  const int lane = tid & 31;
  const int wave = tid >> 5;
  const int ntN = N >> 4;
  const int ktiles = K >> 5;
  for (int tn = wave; tn < ntN; tn += 8) {
    v8f acc = {};
    for (int tk = 0; tk < ktiles; ++tk) {
      v16bf a = load_a_frag(A + tk * 32, lda, lane);
      v16bf b = load_b_frag(Bp + ((size_t)tk * ntN + tn) * 512, lane);
      acc = __builtin_amdgcn_wmma_f32_16x16x32_bf16(false, a, false, b, (short)0, acc, false, false);
    }
    const int hi  = (lane >> 4) & 1;
    const int col = tn * 16 + (lane & 15);
    const float bval = (bias != nullptr && col < biasN) ? bias[col] : 0.0f;
#pragma unroll
    for (int r = 0; r < 8; ++r) {
      int m = r + hi * 8;
      if (m < Bv) out[(size_t)m * rowStride + col] = acc[r] + bval;
    }
  }
}

// ---------- TDM: async 2-D tile load (4 rows x 768 f32, row stride T*768) ----------
#if HAVE_TDM
__device__ __forceinline__ void tdm_load_gi0(const float* gsrc, unsigned lds_off) {
  unsigned long long ga = (unsigned long long)(const void*)gsrc;
  tdm_u32x4 g0;
  g0[0] = 1u;                                           // count=1, user mode
  g0[1] = lds_off;                                      // LDS byte address
  g0[2] = (unsigned)(ga & 0xFFFFFFFFu);                 // global_addr[31:0]
  g0[3] = (unsigned)((ga >> 32) & 0x01FFFFFFu) | 0x80000000u; // addr[56:32] | type=2
  tdm_i32x8 g1;
  g1[0] = 0x00020000;            // data_size=2 (4 bytes), no multicast/pad/iterate
  g1[1] = (int)(768u << 16);     // tensor_dim0 = 768 (bits 79:48 low half)
  g1[2] = (int)(4u << 16);       // tensor_dim0 hi=0 | tensor_dim1 = 4
  g1[3] = (int)(768u << 16);     // tensor_dim1 hi=0 | tile_dim0 = 768
  g1[4] = 4;                     // tile_dim1 = 4, tile_dim2 = 0
  g1[5] = (int)(Tv * 768);       // tensor_dim0_stride = 196608 (bits 207:160)
  g1[6] = 0;                     // stride hi | tensor_dim1_stride lo
  g1[7] = 0;
  tdm_i32x4 z4 = {0, 0, 0, 0};
#if __clang_major__ >= 23
  tdm_i32x8 z8 = {0, 0, 0, 0, 0, 0, 0, 0};
  __builtin_amdgcn_tensor_load_to_lds(g0, g1, z4, z4, z8, 0);
#else
  __builtin_amdgcn_tensor_load_to_lds(g0, g1, z4, z4, 0);
#endif
}
#endif

struct ScanParams {
  const int* cids;
  const float* cat_emb;
  const float* gi0g;      // precomputed li0@Wih0+bih0  [B*T,768]
  const float* xw0g;      // precomputed li0@Wx0+bx0    [B*T,256]
  const ushort_t* Wth0b; const ushort_t* Wth1b;
  const ushort_t* Whh0b; const ushort_t* Whh1b;
  const ushort_t* Wh0b;  const ushort_t* Wh1b;
  const ushort_t* Wih1b; const ushort_t* Wx1b;
  const ushort_t* Wctxb;                      // packed [512x512]
  const float* bth0; const float* bth1;
  const float* bhh0; const float* bhh1;
  const float* bh0;  const float* bh1;
  const float* v0;   const float* v1;
  const float* bv0;  const float* bv1;
  const float* bih1; const float* bx1; const float* bctx;
  float* tape0; float* tape1;                 // [B,T,H] f32
  float* tw0;   float* tw1;                   // tapeWh  [B,T,A] f32
  ushort_t* ctxbuf;                           // [B*T,512] bf16 (cols>=500 zero)
  float* hout;                                // d_out tail: h0 then h1
};

// gilds(12288) + sAbf(16384) + floats(53248)
#define SCAN_SMEM (12288 + 16 * 512 * 2 + (3072 + 3072 + 1024 + 1024 + 1024 + 1024 + 1024 + 2048) * 4)

__global__ void __launch_bounds__(256) k_scan(ScanParams p) {
  extern __shared__ unsigned char smem[];
  float* gilds = (float*)smem;                          // [4][768] TDM destination
  ushort_t* sAbf = (ushort_t*)(smem + 12288);           // [16][512] bf16, rows 4..15 zero
  float* fb    = (float*)(smem + 12288 + 16384);
  float* ghb   = fb;                                    // [4][768]
  float* gib   = ghb + 3072;                            // [4][768]
  float* xwb   = gib + 3072;                            // [4][256]
  float* hWthb = xwb + 1024;                            // [4][256]
  float* aggf  = hWthb + 1024;                          // [4][256]
  float* nhf   = aggf + 1024;                           // [4][256]
  float* scb   = nhf + 1024;                            // [4][256]
  float* hst   = scb + 1024;                            // [2][4][256]

  const int tid  = threadIdx.x;
  const int lane = tid & 31;
  const int wave = tid >> 5;
#if HAVE_TDM
  const unsigned gilds_off = (unsigned)(unsigned long long)(void*)gilds & 0xFFFFFu;
#endif

  for (int i = tid; i < 16 * 512; i += 256) sAbf[i] = 0;
  for (int i = tid; i < 2 * 1024; i += 256) hst[i] = 0.0f;
  __syncthreads();

  for (int t = 0; t < Tv; ++t) {
#if HAVE_TDM
    // async-stage this step's gi0 rows (4 x 768 f32, stride T*768) into LDS
    if (wave == 0) tdm_load_gi0(p.gi0g + (size_t)t * 768, gilds_off);
#endif
    for (int li = 0; li < 2; ++li) {
      const ushort_t* Wth = li ? p.Wth1b : p.Wth0b;
      const ushort_t* Whh = li ? p.Whh1b : p.Whh0b;
      const ushort_t* Wh  = li ? p.Wh1b  : p.Wh0b;
      const float* bth = li ? p.bth1 : p.bth0;
      const float* bhh = li ? p.bhh1 : p.bhh0;
      const float* bh  = li ? p.bh1  : p.bh0;
      const float* vv  = li ? p.v1   : p.v0;
      const float  bvs = li ? p.bv1[0] : p.bv0[0];
      float* tape = li ? p.tape1 : p.tape0;
      float* tw   = li ? p.tw1   : p.tw0;

      if (li == 1) {
        // li1 = relu(nh0 + ce); its two GEMMs (Wih1, Wx1)
        for (int i = tid; i < 1024; i += 256) {
          int b = i >> 8, e = i & 255;
          int cid = p.cids[b * Tv + t];
          float x = nhf[i] + p.cat_emb[(size_t)cid * Ev + e];
          sAbf[b * 512 + e] = f2bf(fmaxf(x, 0.0f));
        }
        __syncthreads();
        scan_gemm(sAbf, 512, p.Wih1b, 768, 256, p.bih1, 768, gib, 768, tid);
        scan_gemm(sAbf, 512, p.Wx1b,  256, 256, p.bx1,  256, xwb, 256, tid);
        __syncthreads();
      }

      // hWth = h @ Wth + bth
      for (int i = tid; i < 1024; i += 256) {
        int b = i >> 8, hh = i & 255;
        sAbf[b * 512 + hh] = f2bf(hst[li * 1024 + i]);
      }
      __syncthreads();
      scan_gemm(sAbf, 512, Wth, 256, 256, bth, 256, hWthb, 256, tid);
      __syncthreads();

      // attention scores over j < t  (tapeWh maintained incrementally)
      for (int idx = wave; idx < 4 * t; idx += 8) {
        int b = idx & 3, j = idx >> 2;
        const float* twr = tw + ((size_t)b * Tv + j) * Av;
        float s = 0.0f;
        for (int a0 = lane; a0 < Av; a0 += 32) {
          float xwv = (li == 0) ? p.xw0g[((size_t)(b * Tv + t)) * Av + a0]
                                : xwb[b * Av + a0];
          float e = tanhf(twr[a0] + bh[a0] + xwv + hWthb[b * Av + a0]);
          s += e * vv[a0];
        }
        for (int off = 16; off; off >>= 1) s += __shfl_down(s, off, 32);
        if (lane == 0) scb[b * Tv + j] = s + bvs;
      }
      __syncthreads();
      // softmax per batch row
      if (tid < 4 && t > 0) {
        float mx = -1e30f;
        for (int j = 0; j < t; ++j) mx = fmaxf(mx, scb[tid * Tv + j]);
        float sum = 0.0f;
        for (int j = 0; j < t; ++j) {
          float e = __expf(scb[tid * Tv + j] - mx);
          scb[tid * Tv + j] = e; sum += e;
        }
        float inv = 1.0f / sum;
        for (int j = 0; j < t; ++j) scb[tid * Tv + j] *= inv;
      }
      __syncthreads();
      // agg = sum_j w[b,j] * tape[b,j,:]  (or h at t==0)
      for (int i = tid; i < 1024; i += 256) {
        int b = i >> 8, hh = i & 255;
        float s = 0.0f;
        const float* tp = tape + (size_t)b * Tv * Hv;
        for (int j = 0; j < t; ++j) s += scb[b * Tv + j] * tp[(size_t)j * Hv + hh];
        aggf[i] = (t > 0) ? s : hst[li * 1024 + i];
      }
      __syncthreads();

      // gh = agg @ Whh + bhh
      for (int i = tid; i < 1024; i += 256) {
        int b = i >> 8, hh = i & 255;
        sAbf[b * 512 + hh] = f2bf(aggf[i]);
      }
      __syncthreads();
      scan_gemm(sAbf, 512, Whh, 768, 256, bhh, 768, ghb, 768, tid);
#if HAVE_TDM
      if (li == 0) __builtin_amdgcn_s_wait_tensorcnt(0);  // gi0 tile must be in LDS
#endif
      __syncthreads();

      // GRU gates
      for (int i = tid; i < 1024; i += 256) {
        int b = i >> 8, hh = i & 255;
        float gir, giu, gin;
        if (li == 0) {
#if HAVE_TDM
          const float* g = gilds + b * 768;
#else
          const float* g = p.gi0g + ((size_t)(b * Tv + t)) * 768;
#endif
          gir = g[hh]; giu = g[256 + hh]; gin = g[512 + hh];
        } else {
          gir = gib[b * 768 + hh]; giu = gib[b * 768 + 256 + hh]; gin = gib[b * 768 + 512 + hh];
        }
        float ghr = ghb[b * 768 + hh], ghu = ghb[b * 768 + 256 + hh], ghn = ghb[b * 768 + 512 + hh];
        float r = 1.0f / (1.0f + __expf(-(gir + ghr)));
        float u = 1.0f / (1.0f + __expf(-(giu + ghu)));
        float n = tanhf(gin + r * ghn);
        float nv = (1.0f - u) * n + u * aggf[i];
        nhf[i] = nv;
        hst[li * 1024 + i] = nv;
        tape[(size_t)b * Tv * Hv + (size_t)t * Hv + hh] = nv;
      }
      __threadfence_block();
      __syncthreads();

      // append tapeWh[:,t,:] = nh @ Wh  (bh folded into attention)
      for (int i = tid; i < 1024; i += 256) {
        int b = i >> 8, hh = i & 255;
        sAbf[b * 512 + hh] = f2bf(nhf[i]);
      }
      __syncthreads();
      scan_gemm(sAbf, 512, Wh, 256, 256, nullptr, 0,
                tw + (size_t)t * Av, Tv * Av, tid);
      __threadfence_block();
      __syncthreads();
    }

    // ctx = tanh(concat(nh1, ce) @ Wctx + bctx); store bf16, K-padded to 512
    for (int i = tid; i < 1024; i += 256) {
      int b = i >> 8, e = i & 255;
      int cid = p.cids[b * Tv + t];
      sAbf[b * 512 + 256 + e] = f2bf(p.cat_emb[(size_t)cid * Ev + e]);
    }
    __syncthreads();
    scan_gemm(sAbf, 512, p.Wctxb, 512, 512, p.bctx, CTXN, ghb, 512, tid);
    __syncthreads();
    for (int i = tid; i < 4 * 512; i += 256) {
      int b = i >> 9, n = i & 511;
      float v = (n < CTXN) ? tanhf(ghb[b * 512 + n]) : 0.0f;
      p.ctxbuf[((size_t)(b * Tv + t)) * CTXP + n] = f2bf(v);
    }
    __threadfence_block();
    __syncthreads();
  }

  for (int i = tid; i < 1024; i += 256) {
    p.hout[i]        = hst[i];
    p.hout[1024 + i] = hst[1024 + i];
  }
}

// ---------------------------------------------------------------------------
extern "C" void kernel_launch(void* const* d_in, const int* in_sizes, int n_in,
                              void* d_out, int out_size, void* d_ws, size_t ws_size,
                              hipStream_t stream) {
  const int*   ids  = (const int*)d_in[0];
  const int*   cids = (const int*)d_in[1];
  const float* emb  = (const float*)d_in[2];
  const float* cemb = (const float*)d_in[3];
  auto L = [&](int li, int k) { return (const float*)d_in[4 + li * 12 + k]; };
  const float* Wctx = (const float*)d_in[28];
  const float* bctx = (const float*)d_in[29];
  const float* Wout = (const float*)d_in[30];
  const float* bout = (const float*)d_in[31];

  size_t off = 0;
  auto alloc = [&](size_t bytes) -> void* {
    void* pp = (char*)d_ws + off;
    off += (bytes + 255) & ~(size_t)255;
    return pp;
  };

  ushort_t* liA    = (ushort_t*)alloc((size_t)Bv * Tv * 512 * 2);
  float*    gi0g   = (float*)alloc((size_t)Bv * Tv * 768 * 4);
  float*    xw0g   = (float*)alloc((size_t)Bv * Tv * 256 * 4);
  float*    tape0  = (float*)alloc((size_t)Bv * Tv * Hv * 4);
  float*    tape1  = (float*)alloc((size_t)Bv * Tv * Hv * 4);
  float*    tw0    = (float*)alloc((size_t)Bv * Tv * Av * 4);
  float*    tw1    = (float*)alloc((size_t)Bv * Tv * Av * 4);
  ushort_t* ctxbuf = (ushort_t*)alloc((size_t)Bv * Tv * CTXP * 2);
  ushort_t* Wih0b  = (ushort_t*)alloc((size_t)512 * 768 * 2);
  ushort_t* Wx0b   = (ushort_t*)alloc((size_t)512 * 256 * 2);
  ushort_t* Whh0b  = (ushort_t*)alloc((size_t)256 * 768 * 2);
  ushort_t* Wh0b   = (ushort_t*)alloc((size_t)256 * 256 * 2);
  ushort_t* Wth0b  = (ushort_t*)alloc((size_t)256 * 256 * 2);
  ushort_t* Wih1b  = (ushort_t*)alloc((size_t)256 * 768 * 2);
  ushort_t* Whh1b  = (ushort_t*)alloc((size_t)256 * 768 * 2);
  ushort_t* Wh1b   = (ushort_t*)alloc((size_t)256 * 256 * 2);
  ushort_t* Wx1b   = (ushort_t*)alloc((size_t)256 * 256 * 2);
  ushort_t* Wth1b  = (ushort_t*)alloc((size_t)256 * 256 * 2);
  ushort_t* Wctxb  = (ushort_t*)alloc((size_t)512 * 512 * 2);
  ushort_t* Woutb  = (ushort_t*)alloc((size_t)512 * Vv * 2);

  auto pack = [&](const float* src, ushort_t* dst, int sR, int sC, int Kp, int Np) {
    long n = (long)Kp * Np;
    k_pack_b<<<(unsigned)((n + 255) / 256), 256, 0, stream>>>(src, dst, sR, sC, Kp, Np);
  };
  pack(L(0, 0), Wih0b, 512, 768, 512, 768);
  pack(L(0, 6), Wx0b,  512, 256, 512, 256);
  pack(L(0, 2), Whh0b, 256, 768, 256, 768);
  pack(L(0, 4), Wh0b,  256, 256, 256, 256);
  pack(L(0, 8), Wth0b, 256, 256, 256, 256);
  pack(L(1, 0), Wih1b, 256, 768, 256, 768);
  pack(L(1, 2), Whh1b, 256, 768, 256, 768);
  pack(L(1, 4), Wh1b,  256, 256, 256, 256);
  pack(L(1, 6), Wx1b,  256, 256, 256, 256);
  pack(L(1, 8), Wth1b, 256, 256, 256, 256);
  pack(Wctx,    Wctxb, 512, CTXN, 512, 512);   // pad N 500 -> 512
  pack(Wout,    Woutb, CTXN, Vv,  512, Vv);    // pad K 500 -> 512 (zero rows)

  k_prep<<<2048, 256, 0, stream>>>(ids, cids, emb, cemb, liA);

  // gi0 = li0 @ Wih0 + bih0 ; xw0 = li0 @ Wx0 + bx0   (batched WMMA GEMMs)
  k_gemm_bf16<<<64 * (48 / 4) / 8, 256, 0, stream>>>(liA, Wih0b, L(0, 1), gi0g,
                                                     1024, 768, 512, 512, 768, 768);
  k_gemm_bf16<<<64 * (16 / 4) / 8, 256, 0, stream>>>(liA, Wx0b, L(0, 7), xw0g,
                                                     1024, 256, 512, 512, 256, 256);

  ScanParams sp;
  sp.cids = cids; sp.cat_emb = cemb; sp.gi0g = gi0g; sp.xw0g = xw0g;
  sp.Wth0b = Wth0b; sp.Wth1b = Wth1b; sp.Whh0b = Whh0b; sp.Whh1b = Whh1b;
  sp.Wh0b = Wh0b;   sp.Wh1b = Wh1b;   sp.Wih1b = Wih1b; sp.Wx1b = Wx1b;
  sp.Wctxb = Wctxb;
  sp.bth0 = L(0, 9);  sp.bth1 = L(1, 9);
  sp.bhh0 = L(0, 3);  sp.bhh1 = L(1, 3);
  sp.bh0  = L(0, 5);  sp.bh1  = L(1, 5);
  sp.v0   = L(0, 10); sp.v1   = L(1, 10);
  sp.bv0  = L(0, 11); sp.bv1  = L(1, 11);
  sp.bih1 = L(1, 1);  sp.bx1  = L(1, 7);  sp.bctx = bctx;
  sp.tape0 = tape0; sp.tape1 = tape1; sp.tw0 = tw0; sp.tw1 = tw1;
  sp.ctxbuf = ctxbuf;
  sp.hout = (float*)d_out + (size_t)Bv * Tv * Vv;

  k_scan<<<1, 256, SCAN_SMEM, stream>>>(sp);

  // logits = ctx @ Wout + bout : [1024,512] x [512,32000] bf16 WMMA
  k_gemm_bf16<<<64 * (2000 / 4) / 8, 256, 0, stream>>>(ctxbuf, Woutb, bout, (float*)d_out,
                                                       1024, Vv, 512, 512, Vv, Vv);
}
